// Attention_42202348650800
// MI455X (gfx1250) — matrix operand order repaired and checked by reading, not compiled, not run on hardware
//
#include <hip/hip_runtime.h>
#include <hip/hip_bf16.h>

typedef __attribute__((ext_vector_type(8)))  _Float16 v8h;
typedef __attribute__((ext_vector_type(16))) _Float16 v16h;
typedef __attribute__((ext_vector_type(8)))  float    v8f;

#define B_   2
#define T_   2048
#define HID_ 1024
#define H_   16
#define D_   64

static __device__ __forceinline__ v16h cat8(v8h a, v8h b) {
    return __builtin_shufflevector(a, b, 0,1,2,3,4,5,6,7,8,9,10,11,12,13,14,15);
}

// ---------------------------------------------------------------- convert
__global__ __launch_bounds__(256) void f32_to_f16_k(const float* __restrict__ in,
                                                    _Float16* __restrict__ out, int n) {
    int i = blockIdx.x * blockDim.x + threadIdx.x;
    if (i < n) out[i] = (_Float16)in[i];
}

// ---------------------------------------------------------------- GEMM
// C[M,N] (f32) = A[M,K] (f16, row-major) * W[N,K]^T (f16, row-major)
// One wave computes a 32(M) x 64(N) tile: 8 accumulators; every A tile is
// reused 4x, every B tile 2x. All loads of a k-step are issued before the
// WMMA chain so they overlap with matrix ops (partial loadcnt waits).
__global__ __launch_bounds__(256) void gemm_f16_wmma(const _Float16* __restrict__ A,
                                                     const _Float16* __restrict__ W,
                                                     float* __restrict__ C,
                                                     int M, int N, int K) {
    const int wave  = blockIdx.x * 8 + (threadIdx.x >> 5);
    const int lane  = threadIdx.x & 31;
    const int half  = lane >> 4;
    const int l16   = lane & 15;
    const int nT    = N >> 6;            // 64-wide N tiles
    const int tm    = wave / nT;         // 32-row tile
    const int tn    = wave % nT;
    const int ncol  = tn * 64 + l16;

    v8f acc[2][4] = {};
    const _Float16* ap0 = A + (size_t)(tm * 32 + l16) * K;       // rows 0..15
    const _Float16* ap1 = ap0 + (size_t)16 * K;                  // rows 16..31
    const _Float16* bp  = W + (size_t)ncol * K + half * 16;

    for (int kk = 0; kk < K; kk += 32) {
        // prefetch next k-chunk into the cache hierarchy
        if (kk + 32 < K) {
            __builtin_prefetch(ap0 + kk + 32 + half * 8, 0, 3);
            __builtin_prefetch(ap1 + kk + 32 + half * 8, 0, 3);
            __builtin_prefetch(bp + kk + 32, 0, 3);
        }
        // A 16x32 f16 layout: 16 contiguous K at half*16
        v16h a0 = *(const v16h*)(ap0 + kk + half * 16);
        v16h a1 = *(const v16h*)(ap1 + kk + half * 16);
        // B 32x16 f16 layout: lane n = col, 16 contiguous K at half*16
        v16h b0 = *(const v16h*)(bp + (size_t)0  * K + kk);
        v16h b1 = *(const v16h*)(bp + (size_t)16 * K + kk);
        v16h b2 = *(const v16h*)(bp + (size_t)32 * K + kk);
        v16h b3 = *(const v16h*)(bp + (size_t)48 * K + kk);
        acc[0][0] = __builtin_amdgcn_wmma_f32_16x16x32_f16(false, a0, false, b0, (short)0, acc[0][0], false, false);
        acc[0][1] = __builtin_amdgcn_wmma_f32_16x16x32_f16(false, a0, false, b1, (short)0, acc[0][1], false, false);
        acc[0][2] = __builtin_amdgcn_wmma_f32_16x16x32_f16(false, a0, false, b2, (short)0, acc[0][2], false, false);
        acc[0][3] = __builtin_amdgcn_wmma_f32_16x16x32_f16(false, a0, false, b3, (short)0, acc[0][3], false, false);
        acc[1][0] = __builtin_amdgcn_wmma_f32_16x16x32_f16(false, a1, false, b0, (short)0, acc[1][0], false, false);
        acc[1][1] = __builtin_amdgcn_wmma_f32_16x16x32_f16(false, a1, false, b1, (short)0, acc[1][1], false, false);
        acc[1][2] = __builtin_amdgcn_wmma_f32_16x16x32_f16(false, a1, false, b2, (short)0, acc[1][2], false, false);
        acc[1][3] = __builtin_amdgcn_wmma_f32_16x16x32_f16(false, a1, false, b3, (short)0, acc[1][3], false, false);
    }
    // C layout: VGPR r -> row base + r + 8*half, col = ncol (+16*t)
    #pragma unroll
    for (int mi = 0; mi < 2; ++mi)
        #pragma unroll
        for (int t = 0; t < 4; ++t)
            #pragma unroll
            for (int r = 0; r < 8; ++r)
                C[(size_t)(tm * 32 + mi * 16 + r + 8 * half) * N + ncol + t * 16]
                    = acc[mi][t][r];
}

// ---------------------------------------------------------------- RoPE + repack
// qkv f32 [B*T, 3072] -> q16/k16 [B,T,H,D] (RoPE'd), vT [B,H,D,T]
__global__ __launch_bounds__(256) void rope_pack_k(const float* __restrict__ qkv,
                                                   const float* __restrict__ cosT,
                                                   const float* __restrict__ sinT,
                                                   _Float16* __restrict__ q16,
                                                   _Float16* __restrict__ k16,
                                                   _Float16* __restrict__ vT) {
    int i = blockIdx.x * blockDim.x + threadIdx.x;   // i = ((b*T+t)*H + h)*D + d
    int d = i & 63;
    int h = (i >> 6) & 15;
    int t = (i >> 10) & (T_ - 1);
    int b = i >> 21;
    const float* base = qkv + (size_t)(b * T_ + t) * 3072;
    float c = cosT[t * 64 + d], s = sinT[t * 64 + d];
    int   dp  = (d < 32) ? d + 32 : d - 32;
    float sgn = (d < 32) ? -1.0f : 1.0f;
    float qv = base[h * 64 + d],        qp = base[h * 64 + dp];
    float kv = base[1024 + h * 64 + d], kp = base[1024 + h * 64 + dp];
    float vv = base[2048 + h * 64 + d];
    q16[i] = (_Float16)(qv * c + sgn * qp * s);
    k16[i] = (_Float16)(kv * c + sgn * kp * s);
    vT[(((size_t)b * H_ + h) * D_ + d) * T_ + t] = (_Float16)vv;
}

// ---------------------------------------------------------------- flash attention
// One wave: 16 queries of one (b,h); loops over keys in 32-wide blocks.
__global__ __launch_bounds__(256) void flash_attn_k(const _Float16* __restrict__ q16,
                                                    const _Float16* __restrict__ k16,
                                                    const _Float16* __restrict__ vT,
                                                    _Float16* __restrict__ ctx) {
    __shared__ alignas(32) _Float16 lds[8 * 16 * 32];   // 1KB per wave
    const int wave = blockIdx.x * 8 + (threadIdx.x >> 5);
    const int wIb  = threadIdx.x >> 5;
    const int lane = threadIdx.x & 31;
    const int half = lane >> 4;
    const int l16  = lane & 15;
    const int qb   = wave & 127;          // T/16
    const int h    = (wave >> 7) & 15;
    const int b    = wave >> 11;

    // Q tile (16x64) as two 16x32 A-matrices
    const _Float16* qbase = q16 + (((size_t)(b * T_) + qb * 16 + l16) * H_ + h) * D_;
    v16h aq[2];
    #pragma unroll
    for (int kk = 0; kk < 2; ++kk)
        aq[kk] = *(const v16h*)(qbase + kk * 32 + half * 16);

    float mrow[8], lrow[8];
    v8f acc[4] = {};
    #pragma unroll
    for (int r = 0; r < 8; ++r) { mrow[r] = -3.0e38f; lrow[r] = 0.0f; }

    const int   qmax  = qb * 16 + 15;
    const float scale = 0.125f;           // 1/sqrt(64)
    _Float16* plds = lds + wIb * 512;

    // loop-invariant per-lane bases
    const _Float16* kbase = k16 + (((size_t)(b * T_) + l16) * H_ + h) * D_ + half * 16;
    const _Float16* vbase = vT + (((size_t)b * H_ + h) * D_ + l16) * T_ + half * 16;

    for (int kb = 0; kb <= qmax; kb += 32) {
        // prefetch next key block (K rows and V columns)
        if (kb + 32 <= qmax) {
            __builtin_prefetch(kbase + (size_t)(kb + 32) * (H_ * D_), 0, 3);
            __builtin_prefetch(kbase + (size_t)(kb + 48) * (H_ * D_), 0, 3);
            __builtin_prefetch(vbase + kb + 32, 0, 3);
        }
        // ---- load all four K-tiles, then run the S WMMA chain
        const _Float16* kp0 = kbase + (size_t)kb * (H_ * D_);
        const _Float16* kp1 = kbase + (size_t)(kb + 16) * (H_ * D_);
        v16h bk00 = *(const v16h*)(kp0);
        v16h bk01 = *(const v16h*)(kp0 + 32);
        v16h bk10 = *(const v16h*)(kp1);
        v16h bk11 = *(const v16h*)(kp1 + 32);
        v8f s[2] = {};
        s[0] = __builtin_amdgcn_wmma_f32_16x16x32_f16(false, aq[0], false, bk00, (short)0, s[0], false, false);
        s[1] = __builtin_amdgcn_wmma_f32_16x16x32_f16(false, aq[0], false, bk10, (short)0, s[1], false, false);
        s[0] = __builtin_amdgcn_wmma_f32_16x16x32_f16(false, aq[1], false, bk01, (short)0, s[0], false, false);
        s[1] = __builtin_amdgcn_wmma_f32_16x16x32_f16(false, aq[1], false, bk11, (short)0, s[1], false, false);

        // ---- scale + causal mask
        #pragma unroll
        for (int t = 0; t < 2; ++t) {
            const int key = kb + t * 16 + l16;
            #pragma unroll
            for (int r = 0; r < 8; ++r) {
                const int qi = qb * 16 + r + 8 * half;
                s[t][r] = (key <= qi) ? s[t][r] * scale : -1.0e9f;
            }
        }
        // ---- online softmax: row max (rows live across the 16 lanes of a half)
        float corr[8];
        #pragma unroll
        for (int r = 0; r < 8; ++r) {
            float m = fmaxf(s[0][r], s[1][r]);
            m = fmaxf(m, __shfl_xor(m, 1, 16));
            m = fmaxf(m, __shfl_xor(m, 2, 16));
            m = fmaxf(m, __shfl_xor(m, 4, 16));
            m = fmaxf(m, __shfl_xor(m, 8, 16));
            float nm = fmaxf(mrow[r], m);
            corr[r]  = __expf(mrow[r] - nm);
            mrow[r]  = nm;
        }
        // ---- exp, stage P to LDS (C-layout -> row-major for A-layout reload)
        #pragma unroll
        for (int t = 0; t < 2; ++t)
            #pragma unroll
            for (int r = 0; r < 8; ++r) {
                float p = __expf(s[t][r] - mrow[r]);
                s[t][r] = p;
                plds[(r + 8 * half) * 32 + t * 16 + l16] = (_Float16)p;
            }
        // ---- row sums + accumulator rescale
        #pragma unroll
        for (int r = 0; r < 8; ++r) {
            float rs = s[0][r] + s[1][r];
            rs += __shfl_xor(rs, 1, 16);
            rs += __shfl_xor(rs, 2, 16);
            rs += __shfl_xor(rs, 4, 16);
            rs += __shfl_xor(rs, 8, 16);
            lrow[r] = lrow[r] * corr[r] + rs;
            #pragma unroll
            for (int ct = 0; ct < 4; ++ct) acc[ct][r] *= corr[r];
        }
        // ---- reload P as 16x32 A-matrix (same-wave LDS is in-order)
        asm volatile("s_wait_dscnt 0x0" ::: "memory");
        v16h ap = *(const v16h*)(plds + l16 * 32 + half * 16);
        // ---- load all four V-tiles, then run the PV WMMA chain
        v16h bv0 = *(const v16h*)(vbase + (size_t)0  * T_ + kb);
        v16h bv1 = *(const v16h*)(vbase + (size_t)16 * T_ + kb);
        v16h bv2 = *(const v16h*)(vbase + (size_t)32 * T_ + kb);
        v16h bv3 = *(const v16h*)(vbase + (size_t)48 * T_ + kb);
        acc[0] = __builtin_amdgcn_wmma_f32_16x16x32_f16(false, ap, false, bv0, (short)0, acc[0], false, false);
        acc[1] = __builtin_amdgcn_wmma_f32_16x16x32_f16(false, ap, false, bv1, (short)0, acc[1], false, false);
        acc[2] = __builtin_amdgcn_wmma_f32_16x16x32_f16(false, ap, false, bv2, (short)0, acc[2], false, false);
        acc[3] = __builtin_amdgcn_wmma_f32_16x16x32_f16(false, ap, false, bv3, (short)0, acc[3], false, false);
    }
    // ---- normalize and store context [B,T,H*D] as f16
    #pragma unroll
    for (int ct = 0; ct < 4; ++ct)
        #pragma unroll
        for (int r = 0; r < 8; ++r) {
            float o = acc[ct][r] / lrow[r];
            ctx[((size_t)(b * T_) + qb * 16 + r + 8 * half) * (H_ * D_)
                + h * D_ + ct * 16 + l16] = (_Float16)o;
        }
}

// ---------------------------------------------------------------- launch
extern "C" void kernel_launch(void* const* d_in, const int* in_sizes, int n_in,
                              void* d_out, int out_size, void* d_ws, size_t ws_size,
                              hipStream_t stream) {
    const float* hidden = (const float*)d_in[0];   // [B,T,1024]
    const float* cosT   = (const float*)d_in[1];   // [T,64]
    const float* sinT   = (const float*)d_in[2];   // [T,64]
    const float* w_qkv  = (const float*)d_in[3];   // [3072,1024]
    const float* w_o    = (const float*)d_in[4];   // [1024,1024]

    char* ws = (char*)d_ws;
    _Float16* h16    = (_Float16*)(ws + 0);          //  8 MB
    _Float16* wqkv16 = (_Float16*)(ws + 8388608);    //  6 MB
    _Float16* wo16   = (_Float16*)(ws + 14680064);   //  2 MB
    float*    qkv32  = (float*)   (ws + 16777216);   // 48 MB
    _Float16* q16    = (_Float16*)(ws + 67108864);   //  8 MB
    _Float16* k16    = (_Float16*)(ws + 75497472);   //  8 MB
    _Float16* vT16   = (_Float16*)(ws + 83886080);   //  8 MB
    _Float16* ctx16  = (_Float16*)(ws + 92274688);   //  8 MB

    const int nH = B_ * T_ * HID_;        // 4,194,304
    const int nW = 3072 * 1024;           // 3,145,728
    const int nO = 1024 * 1024;           // 1,048,576

    f32_to_f16_k<<<(nH + 255) / 256, 256, 0, stream>>>(hidden, h16, nH);
    f32_to_f16_k<<<(nW + 255) / 256, 256, 0, stream>>>(w_qkv, wqkv16, nW);
    f32_to_f16_k<<<(nO + 255) / 256, 256, 0, stream>>>(w_o, wo16, nO);

    // QKV projection: M=4096, N=3072, K=1024 -> (128 * 48) waves / 8 = 768 blocks
    gemm_f16_wmma<<<768, 256, 0, stream>>>(h16, wqkv16, qkv32, 4096, 3072, 1024);

    // RoPE + repack: one thread per (b,t,h,d)
    rope_pack_k<<<nH / 256, 256, 0, stream>>>(qkv32, cosT, sinT, q16, k16, vT16);

    // Flash attention: B*H*(T/16) = 4096 waves / 8 = 512 blocks
    flash_attn_k<<<512, 256, 0, stream>>>(q16, k16, vT16, ctx16);

    // Output projection: M=4096, N=1024, K=1024 -> (128 * 16) waves / 8 = 256 blocks
    gemm_f16_wmma<<<256, 256, 0, stream>>>(ctx16, wo16, (float*)d_out, 4096, 1024, 1024);
}
